// DigitConvolutionalModel_44169443672332
// MI455X (gfx1250) — matrix operand order repaired and checked by reading, not compiled, not run on hardware
//
#include <hip/hip_runtime.h>
#include <hip/hip_bf16.h>

typedef _Float16 v16h __attribute__((ext_vector_type(16)));
typedef _Float16 v8h  __attribute__((ext_vector_type(8)));
typedef float    v8f  __attribute__((ext_vector_type(8)));

union AFrag { v16h v; v8h h[2]; };

#define K1P 704   // 676 padded to multiple of 32
#define KS1 22    // 704/32
#define KS2 8     // 256/32
#define KS3 4     // 128/32

// ---------------------------------------------------------------------------
// Prep: convert W[N][K] (fp32, row-major, out-major like w @ x.T usage) into
// fragment-ready f16 layout: [ntile][kstep][lane 0..31][16 halfs], where
// lane = (khalf<<4) | n_local and the 16 halfs are two 8-wide K runs:
//   n = tile*16 + (lane&15),  k = step*32 + (idx>>3)*16 + (lane>>4)*8 + (idx&7)
// This matches the CDNA5 B-matrix (32x16, 16-bit) VGPR layout, so the main
// kernel loads a whole B fragment with one 32B lane-contiguous load.
// ---------------------------------------------------------------------------
__global__ __launch_bounds__(256) void swizzle_kernel(
    const float* __restrict__ w, _Float16* __restrict__ o,
    int N, int K, int ntiles, int ksteps)
{
    int total = ntiles * ksteps * 512;
    for (int g = blockIdx.x * blockDim.x + threadIdx.x; g < total;
         g += gridDim.x * blockDim.x) {
        int t    = g / (ksteps * 512);
        int rem  = g - t * (ksteps * 512);
        int s    = rem >> 9;
        int e    = rem & 511;
        int lane = e >> 4;
        int idx  = e & 15;
        int n = t * 16 + (lane & 15);
        int k = s * 32 + (idx >> 3) * 16 + (lane >> 4) * 8 + (idx & 7);
        float v = (n < N && k < K) ? w[(size_t)n * K + k] : 0.0f;
        o[g] = (_Float16)v;
    }
}

// ---------------------------------------------------------------------------
// Fused conv + 4-layer MLP. One workgroup (8 wave32s) per 16 batch rows.
// ---------------------------------------------------------------------------
__global__ __launch_bounds__(256) void fused_mlp_kernel(
    const float* __restrict__ x,  const float* __restrict__ cw,
    const float* __restrict__ b1, const float* __restrict__ b2,
    const float* __restrict__ b3, const float* __restrict__ w4,
    const float* __restrict__ b4,
    const _Float16* __restrict__ w1s, const _Float16* __restrict__ w2s,
    const _Float16* __restrict__ w3s, float* __restrict__ out)
{
    __shared__ float    xb [16][784];
    __shared__ _Float16 hb [16][K1P];
    __shared__ _Float16 h1b[16][256];
    __shared__ _Float16 h2b[16][128];
    __shared__ _Float16 h3b[16][64];
    __shared__ float    cwl[9];

    const int tid  = threadIdx.x;
    const int lane = tid & 31;
    const int wid  = tid >> 5;
    const int m0   = blockIdx.x * 16;

    // ---- stage x tile (16 x 784 fp32) into LDS, coalesced float4 ----
    {
        const float4* xg = (const float4*)(x + (size_t)m0 * 784);
        float4* xl = (float4*)&xb[0][0];
        for (int v = tid; v < 16 * 196; v += 256) xl[v] = xg[v];
        if (tid < 9) cwl[tid] = cw[tid];
    }
    __syncthreads();

    // ---- conv 3x3 VALID -> hb f16, zero-pad K 676..703 ----
    for (int idx = tid; idx < 16 * 676; idx += 256) {
        int r = idx / 676;
        int p = idx - r * 676;
        int i = p / 26;
        int j = p - i * 26;
        const float* src = &xb[r][i * 28 + j];
        float s = src[ 0]*cwl[0] + src[ 1]*cwl[1] + src[ 2]*cwl[2]
                + src[28]*cwl[3] + src[29]*cwl[4] + src[30]*cwl[5]
                + src[56]*cwl[6] + src[57]*cwl[7] + src[58]*cwl[8];
        hb[r][p] = (_Float16)s;
    }
    for (int idx = tid; idx < 16 * (K1P - 676); idx += 256) {
        int r = idx / (K1P - 676);
        hb[r][676 + (idx - r * (K1P - 676))] = (_Float16)0.0f;
    }
    __syncthreads();

    const int mm = lane & 15;   // row (A) / col (B,C)
    const int hs = lane >> 4;   // half-wave select

    // ---- layer 1: [16x704] @ [704x256], wave owns tiles 2*wid, 2*wid+1 ----
    {
        const int t0 = wid * 2;
        v8f acc0 = {}; v8f acc1 = {};
        for (int s = 0; s < KS1; ++s) {
            AFrag a;
            a.h[0] = *(const v8h*)&hb[mm][s * 32 +      hs * 8];
            a.h[1] = *(const v8h*)&hb[mm][s * 32 + 16 + hs * 8];
            v16h bfa = *(const v16h*)(w1s + ((size_t)( t0      * KS1 + s) * 32 + lane) * 16);
            v16h bfb = *(const v16h*)(w1s + ((size_t)((t0 + 1) * KS1 + s) * 32 + lane) * 16);
            acc0 = __builtin_amdgcn_wmma_f32_16x16x32_f16(false, a.v, false, bfa, (short)0, acc0, false, false);
            acc1 = __builtin_amdgcn_wmma_f32_16x16x32_f16(false, a.v, false, bfb, (short)0, acc1, false, false);
        }
#pragma unroll
        for (int r = 0; r < 8; ++r) {
            int row = hs * 8 + r;
            int c0 = t0 * 16 + mm;
            int c1 = (t0 + 1) * 16 + mm;
            float v0 = fmaxf(acc0[r] + b1[c0], 0.0f);
            float v1 = fmaxf(acc1[r] + b1[c1], 0.0f);
            h1b[row][c0] = (_Float16)v0;
            h1b[row][c1] = (_Float16)v1;
        }
    }
    __syncthreads();

    // ---- layer 2: [16x256] @ [256x128], 8 tiles / 8 waves ----
    {
        const int t = wid;
        v8f acc = {};
        for (int s = 0; s < KS2; ++s) {
            AFrag a;
            a.h[0] = *(const v8h*)&h1b[mm][s * 32 +      hs * 8];
            a.h[1] = *(const v8h*)&h1b[mm][s * 32 + 16 + hs * 8];
            v16h bf = *(const v16h*)(w2s + ((size_t)(t * KS2 + s) * 32 + lane) * 16);
            acc = __builtin_amdgcn_wmma_f32_16x16x32_f16(false, a.v, false, bf, (short)0, acc, false, false);
        }
#pragma unroll
        for (int r = 0; r < 8; ++r) {
            int row = hs * 8 + r;
            int c = t * 16 + mm;
            h2b[row][c] = (_Float16)fmaxf(acc[r] + b2[c], 0.0f);
        }
    }
    __syncthreads();

    // ---- layer 3: [16x128] @ [128x64], 4 tiles on waves 0..3 ----
    if (wid < 4) {
        const int t = wid;
        v8f acc = {};
        for (int s = 0; s < KS3; ++s) {
            AFrag a;
            a.h[0] = *(const v8h*)&h2b[mm][s * 32 +      hs * 8];
            a.h[1] = *(const v8h*)&h2b[mm][s * 32 + 16 + hs * 8];
            v16h bf = *(const v16h*)(w3s + ((size_t)(t * KS3 + s) * 32 + lane) * 16);
            acc = __builtin_amdgcn_wmma_f32_16x16x32_f16(false, a.v, false, bf, (short)0, acc, false, false);
        }
#pragma unroll
        for (int r = 0; r < 8; ++r) {
            int row = hs * 8 + r;
            int c = t * 16 + mm;
            h3b[row][c] = (_Float16)fmaxf(acc[r] + b3[c], 0.0f);
        }
    }
    __syncthreads();

    // ---- layer 4: [16x64] @ [64x10] in fp32 (tiny), write logits ----
    {
        int row = tid >> 4;   // 0..15
        int col = tid & 15;   // 0..15
        if (col < 10) {
            float s = b4[col];
            const float* wr = w4 + col * 64;
#pragma unroll
            for (int k = 0; k < 64; ++k) s += (float)h3b[row][k] * wr[k];
            out[(size_t)(m0 + row) * 10 + col] = s;
        }
    }
}

extern "C" void kernel_launch(void* const* d_in, const int* in_sizes, int n_in,
                              void* d_out, int out_size, void* d_ws, size_t ws_size,
                              hipStream_t stream) {
    const float* x  = (const float*)d_in[0];
    const float* cw = (const float*)d_in[1];
    const float* w1 = (const float*)d_in[2];
    const float* b1 = (const float*)d_in[3];
    const float* w2 = (const float*)d_in[4];
    const float* b2 = (const float*)d_in[5];
    const float* w3 = (const float*)d_in[6];
    const float* b3 = (const float*)d_in[7];
    const float* w4 = (const float*)d_in[8];
    const float* b4 = (const float*)d_in[9];
    float* out = (float*)d_out;

    _Float16* w1s = (_Float16*)d_ws;                 // 16*22*512 halfs
    _Float16* w2s = w1s + (size_t)16 * KS1 * 512;    //  8* 8*512 halfs
    _Float16* w3s = w2s + (size_t)8  * KS2 * 512;    //  4* 4*512 halfs

    swizzle_kernel<<<176, 256, 0, stream>>>(w1, w1s, 256, 676, 16, KS1);
    swizzle_kernel<<<32,  256, 0, stream>>>(w2, w2s, 128, 256,  8, KS2);
    swizzle_kernel<<<8,   256, 0, stream>>>(w3, w3s,  64, 128,  4, KS3);

    const int Bn = in_sizes[0] / 784;     // 16384
    fused_mlp_kernel<<<Bn / 16, 256, 0, stream>>>(
        x, cw, b1, b2, b3, w4, b4, w1s, w2s, w3s, out);
}
